// QuantizerEMA_23106924052661
// MI455X (gfx1250) — matrix operand order repaired and checked by reading, not compiled, not run on hardware
//
#include <hip/hip_runtime.h>
#include <hip/hip_bf16.h>

// ---------------------------------------------------------------------------
// Types for CDNA5 WMMA (wave32, 16x16x32 bf16 -> f32 accum)
// ---------------------------------------------------------------------------
typedef __attribute__((ext_vector_type(16))) __bf16          v16bf;
typedef __attribute__((ext_vector_type(8)))  float           v8f;
typedef __attribute__((ext_vector_type(8)))  unsigned short  ushort8;
typedef __attribute__((ext_vector_type(4)))  int             v4i;

union BFrag {
  v16bf          v;
  ushort8        h[2];
  unsigned short u[16];
};

static __device__ __forceinline__ unsigned short f32_to_bf16(float f) {
  unsigned u = __float_as_uint(f);
  u += 0x7FFFu + ((u >> 16) & 1u);           // round-to-nearest-even
  return (unsigned short)(u >> 16);
}
static __device__ __forceinline__ float bf16_to_f32(unsigned short h) {
  return __uint_as_float(((unsigned)h) << 16);
}

// Problem dims (fixed by the reference)
#define NVEC   32768     // 32*32*32 latent vectors
#define DDIM   256
#define KCODE  4096
#define HWSZ   1024      // 32*32
#define BSZ    32

// ---------------------------------------------------------------------------
// CDNA5 async global->LDS copy (ASYNCcnt-tracked, bypasses VGPRs)
// ---------------------------------------------------------------------------
#if defined(__has_builtin)
#  if __has_builtin(__builtin_amdgcn_global_load_async_to_lds_b128)
#    define HAVE_ASYNC_LDS 1
#  endif
#endif
#ifndef HAVE_ASYNC_LDS
#  define HAVE_ASYNC_LDS 0
#endif

#define GLOBAL_AS __attribute__((address_space(1)))
#define LDS_AS    __attribute__((address_space(3)))

static __device__ __forceinline__ void wait_async_lds() {
#if HAVE_ASYNC_LDS
#  if __has_builtin(__builtin_amdgcn_s_wait_asynccnt)
  __builtin_amdgcn_s_wait_asynccnt(0);
#  else
  asm volatile("s_wait_asynccnt 0" ::: "memory");
#  endif
#endif
}

// Stage one 32-code chunk (hi+lo bf16, 32 KB) into LDS; 8 x b128 per thread.
static __device__ __forceinline__ void stage_chunk(
    unsigned short* __restrict__ dH, unsigned short* __restrict__ dL,
    const unsigned short* __restrict__ gH, const unsigned short* __restrict__ gL,
    int tid) {
#if HAVE_ASYNC_LDS
#pragma unroll
  for (int i = 0; i < 4; ++i) {
    __builtin_amdgcn_global_load_async_to_lds_b128(
        (GLOBAL_AS v4i*)((const uint4*)gH + tid + i * 256),
        (LDS_AS v4i*)((uint4*)dH + tid + i * 256), 0, 0);
    __builtin_amdgcn_global_load_async_to_lds_b128(
        (GLOBAL_AS v4i*)((const uint4*)gL + tid + i * 256),
        (LDS_AS v4i*)((uint4*)dL + tid + i * 256), 0, 0);
  }
#else
  // Fallback: batch all global loads before any LDS store (one wait, not 8)
  uint4 tH[4], tL[4];
#pragma unroll
  for (int i = 0; i < 4; ++i) {
    tH[i] = ((const uint4*)gH)[tid + i * 256];
    tL[i] = ((const uint4*)gL)[tid + i * 256];
  }
#pragma unroll
  for (int i = 0; i < 4; ++i) {
    ((uint4*)dH)[tid + i * 256] = tH[i];
    ((uint4*)dL)[tid + i * 256] = tL[i];
  }
#endif
}

// ---------------------------------------------------------------------------
// Kernel 1: split codebook into bf16 hi/lo, exact fp32 ||e||^2, zero loss slot
// ---------------------------------------------------------------------------
__global__ void prep_kernel(const float* __restrict__ E,
                            unsigned short* __restrict__ Ehi,
                            unsigned short* __restrict__ Elo,
                            float* __restrict__ Enorm,
                            float* __restrict__ lossSlot) {
  const int k = blockIdx.x;
  const int t = threadIdx.x;
  const float e = E[k * DDIM + t];
  const unsigned short hi = f32_to_bf16(e);
  Ehi[k * DDIM + t] = hi;
  Elo[k * DDIM + t] = f32_to_bf16(e - bf16_to_f32(hi));

  __shared__ float red[256];
  red[t] = e * e;
  __syncthreads();
  for (int s = 128; s > 0; s >>= 1) {
    if (t < s) red[t] += red[t + s];
    __syncthreads();
  }
  if (t == 0) {
    Enorm[k] = red[0];
    if (k == 0) *lossSlot = 0.0f;   // re-zero every launch (deterministic)
  }
}

// ---------------------------------------------------------------------------
// Kernel 2: distance GEMM + argmin via split-bf16 WMMA, double-buffered LDS
// 256 blocks x 256 threads (8 waves). Wave w owns rows blk*128 + w*16 .. +15.
// ---------------------------------------------------------------------------
__global__ __launch_bounds__(256) void argmin_kernel(
    const float* __restrict__ z,
    const unsigned short* __restrict__ Ehi,
    const unsigned short* __restrict__ Elo,
    const float* __restrict__ Enorm,
    int* __restrict__ bestIdx) {
  // [buffer][hi/lo][32 codes x 256 dims] = 64 KB
  __shared__ alignas(16) unsigned short sE[2][2][32 * DDIM];

  const int tid     = threadIdx.x;
  const int lane    = tid & 31;
  const int wave    = tid >> 5;
  const int rowBase = blockIdx.x * 128 + wave * 16;
  const int row     = rowBase + (lane & 15);
  // A-matrix 16-bit layout: lanes 0-15 carry K {0..7,16..23}, lanes 16-31 K {8..15,24..31}
  const int kbA = (lane >> 4) << 3;
  // B-matrix layout: lanes 0-15 carry K {0..7,16..23}, lanes 16-31 K {8..15,24..31}
  const int kbB = (lane >> 4) << 3;

  // Prefetch chunk 0 into buffer 0 while we build A-fragments
  stage_chunk(sE[0][0], sE[0][1], Ehi, Elo, tid);

  // Load this wave's 16 z-rows as resident bf16 hi/lo A-fragments (8 k-steps of 32)
  BFrag ahi[8], alo[8];
#pragma unroll
  for (int c = 0; c < 8; ++c) {
    const float* p = z + row * DDIM + c * 32 + kbA;
    float4 f[4];
    f[0] = *(const float4*)(p);
    f[1] = *(const float4*)(p + 4);
    f[2] = *(const float4*)(p + 16);
    f[3] = *(const float4*)(p + 20);
    const float* fv = (const float*)f;
#pragma unroll
    for (int e = 0; e < 16; ++e) {
      const unsigned short h = f32_to_bf16(fv[e]);
      ahi[c].u[e] = h;
      alo[c].u[e] = f32_to_bf16(fv[e] - bf16_to_f32(h));
    }
  }

  float best[8];
  int   bidx[8];
#pragma unroll
  for (int r = 0; r < 8; ++r) { best[r] = 3.4e38f; bidx[r] = 0; }

  wait_async_lds();
  __syncthreads();

  const int col0 = lane & 15;        // column group 0 (this lane's column)
  const int col1 = 16 + (lane & 15); // column group 1

#pragma unroll 1
  for (int it = 0; it < KCODE / 32; ++it) {
    const int cur = it & 1;
    // Prefetch next chunk into the other buffer (overlaps with WMMA below)
    if (it + 1 < KCODE / 32) {
      stage_chunk(sE[cur ^ 1][0], sE[cur ^ 1][1],
                  Ehi + (it + 1) * 32 * DDIM, Elo + (it + 1) * 32 * DDIM, tid);
    }

    const unsigned short* __restrict__ curH = sE[cur][0];
    const unsigned short* __restrict__ curL = sE[cur][1];

    // Two independent accumulator chains (col groups 0 and 1), interleaved
    v8f acc0 = {}, acc1 = {};
#pragma unroll
    for (int c = 0; c < 8; ++c) {
      BFrag bhi0, blo0, bhi1, blo1;
      const int ko = c * 32 + kbB;
      const ushort8* h0 = (const ushort8*)(curH + col0 * DDIM + ko);
      const ushort8* l0 = (const ushort8*)(curL + col0 * DDIM + ko);
      const ushort8* h1 = (const ushort8*)(curH + col1 * DDIM + ko);
      const ushort8* l1 = (const ushort8*)(curL + col1 * DDIM + ko);
      bhi0.h[0] = h0[0]; bhi0.h[1] = h0[2];
      blo0.h[0] = l0[0]; blo0.h[1] = l0[2];
      bhi1.h[0] = h1[0]; bhi1.h[1] = h1[2];
      blo1.h[0] = l1[0]; blo1.h[1] = l1[2];
      // split-bf16: hi*hi + hi*lo + lo*hi  (fp32 accumulate), chains interleaved
      acc0 = __builtin_amdgcn_wmma_f32_16x16x32_bf16(false, ahi[c].v, false, bhi0.v,
                                                     (short)0, acc0, false, false);
      acc1 = __builtin_amdgcn_wmma_f32_16x16x32_bf16(false, ahi[c].v, false, bhi1.v,
                                                     (short)0, acc1, false, false);
      acc0 = __builtin_amdgcn_wmma_f32_16x16x32_bf16(false, ahi[c].v, false, blo0.v,
                                                     (short)0, acc0, false, false);
      acc1 = __builtin_amdgcn_wmma_f32_16x16x32_bf16(false, ahi[c].v, false, blo1.v,
                                                     (short)0, acc1, false, false);
      acc0 = __builtin_amdgcn_wmma_f32_16x16x32_bf16(false, alo[c].v, false, bhi0.v,
                                                     (short)0, acc0, false, false);
      acc1 = __builtin_amdgcn_wmma_f32_16x16x32_bf16(false, alo[c].v, false, bhi1.v,
                                                     (short)0, acc1, false, false);
    }

    const int colBase = it * 32;
    const float en0 = Enorm[colBase + col0];
    const float en1 = Enorm[colBase + col1];
#pragma unroll
    for (int r = 0; r < 8; ++r) {
      const float d0 = en0 - 2.0f * acc0[r];   // argmin-equivalent distance
      if (d0 < best[r]) { best[r] = d0; bidx[r] = colBase + col0; }
      const float d1 = en1 - 2.0f * acc1[r];
      if (d1 < best[r]) { best[r] = d1; bidx[r] = colBase + col1; }
    }

    wait_async_lds();   // prefetch of next buffer complete (this wave)
    __syncthreads();    // all waves done reading cur + done prefetching next
  }

  // Cross-lane argmin over the 16 lanes sharing a row set
  // (C layout: VGPR r -> M=r for lanes 0-15, M=8+r for lanes 16-31)
#pragma unroll
  for (int m = 1; m <= 8; m <<= 1) {
#pragma unroll
    for (int r = 0; r < 8; ++r) {
      const float ov = __shfl_xor(best[r], m, 32);
      const int   oi = __shfl_xor(bidx[r], m, 32);
      if (ov < best[r] || (ov == best[r] && oi < bidx[r])) {
        best[r] = ov; bidx[r] = oi;
      }
    }
  }
  if ((lane & 15) == 0) {
    const int m0 = rowBase + ((lane >> 4) << 3);
#pragma unroll
    for (int r = 0; r < 8; ++r) bestIdx[m0 + r] = bidx[r];
  }
}

// ---------------------------------------------------------------------------
// Kernel 3: exact fp32 gather + NHWC->NCHW transpose + commitment loss + indices
// grid (32 hw-tiles, 32 batches), block (32, 8)
// ---------------------------------------------------------------------------
__global__ void quantize_kernel(const float* __restrict__ z,
                                const float* __restrict__ E,
                                const int* __restrict__ bestIdx,
                                float* __restrict__ outQ,
                                float* __restrict__ outIdx,
                                float* __restrict__ outLoss) {
  __shared__ float eT[32][257];   // padded: conflict-free transposed reads
  __shared__ float red[256];
  const int b      = blockIdx.y;
  const int hwBase = blockIdx.x * 32;
  const int tx = threadIdx.x;           // 0..31
  const int ty = threadIdx.y;           // 0..7
  const int t  = ty * 32 + tx;          // 0..255

  float lsum = 0.0f;
#pragma unroll 1
  for (int r = 0; r < 32; ++r) {
    const int n   = b * HWSZ + hwBase + r;
    const int idx = bestIdx[n];
    const float e  = E[idx * DDIM + t];     // coalesced within a row
    const float zv = z[n * DDIM + t];       // coalesced
    const float d  = e - zv;
    lsum += d * d;
    eT[r][t] = e;
  }
  if (t < 32) {
    const int n = b * HWSZ + hwBase + t;
    outIdx[n] = (float)bestIdx[n];
  }
  red[t] = lsum;
  __syncthreads();

  // write quantized_vector [B, D, H, W]; per-wave (fixed ty) fully coalesced
#pragma unroll
  for (int k = 0; k < 32; ++k) {
    const int d = ty * 32 + k;
    outQ[(b * DDIM + d) * HWSZ + hwBase + tx] = eT[tx][d];
  }

  for (int s = 128; s > 0; s >>= 1) {
    __syncthreads();
    if (t < s) red[t] += red[t + s];
  }
  if (t == 0) {
    // loss = 0.25 * mean((q - z)^2) ; mean over 32768*256 = 8388608 elements
    atomicAdd(outLoss, red[0] * (0.25f / 8388608.0f));
  }
}

// ---------------------------------------------------------------------------
extern "C" void kernel_launch(void* const* d_in, const int* in_sizes, int n_in,
                              void* d_out, int out_size, void* d_ws, size_t ws_size,
                              hipStream_t stream) {
  const float* z = (const float*)d_in[0];   // [32,32,32,256] f32
  const float* E = (const float*)d_in[1];   // [4096,256]     f32

  char* ws = (char*)d_ws;
  unsigned short* Ehi   = (unsigned short*)(ws);                               // 2 MB
  unsigned short* Elo   = (unsigned short*)(ws + (size_t)KCODE * DDIM * 2);    // 2 MB
  float*          Enorm = (float*)(ws + (size_t)KCODE * DDIM * 4);             // 16 KB
  int*            bIdx  = (int*)  (ws + (size_t)KCODE * DDIM * 4 + KCODE * 4); // 128 KB

  float* outQ    = (float*)d_out;                   // 32*256*32*32 = 8388608
  float* outIdx  = outQ + (size_t)BSZ * DDIM * HWSZ;
  float* outLoss = outIdx + NVEC;

  prep_kernel<<<KCODE, 256, 0, stream>>>(E, Ehi, Elo, Enorm, outLoss);
  argmin_kernel<<<NVEC / 128, 256, 0, stream>>>(z, Ehi, Elo, Enorm, bIdx);
  quantize_kernel<<<dim3(HWSZ / 32, BSZ), dim3(32, 8), 0, stream>>>(
      z, E, bIdx, outQ, outIdx, outLoss);
}
